// SCAM_56358560858871
// MI455X (gfx1250) — matrix operand order repaired and checked by reading, not compiled, not run on hardware
//
#include <hip/hip_runtime.h>

// ---------------------------------------------------------------------------
// SCAM (stereo cross-attention) for MI455X / gfx1250, wave32 + WMMA bf16.
// All GEMMs use v_wmma_f32_16x16x32_bf16 (fp32 accumulate). Attention is a
// single-pass flash kernel (online softmax) so the 4096x4096 score matrix is
// never written to memory. The conv-GEMM stages its K-chunks with
// global_load_async_to_lds_b64 (ASYNCcnt) in a double-buffered pipeline, with
// the im2col permutation folded into the per-lane source addresses so LDS
// holds data directly in WMMA A-fragment order.
// ---------------------------------------------------------------------------

typedef __attribute__((ext_vector_type(16))) __bf16        v16bf;
typedef __attribute__((ext_vector_type(8)))  float         v8f;
typedef __attribute__((ext_vector_type(4)))  unsigned int  u32x4;
typedef __attribute__((ext_vector_type(4)))  float         f32x4;

// Workspace element offsets (unsigned short elements, 2 bytes each).
#define E_LNL 0u           // LN(x_l) bf16, NCHW, 64*128*32*32
#define E_LNR 8388608u     // LN(x_r)
#define E_QL  16777216u    // Q_l  [4096][128] bf16
#define E_QR  17301504u    // Q_r
#define E_VTL 17825792u    // V_l transposed [128][4096] bf16
#define E_VTR 18350080u    // V_r transposed
#define E_FRL 18874368u    // F_r2l [4096][128] bf16
#define E_FLR 19398656u    // F_l2r
#define E_WP  19922944u    // 4 packed conv weights, each [64][128][32] bf16
#define E_W3L 20971520u    // Wl3 packed [2048][128] bf16
#define E_W3R 21233664u    // Wr3 packed

__device__ __forceinline__ unsigned short f2bf(float f) {
  unsigned int u = __float_as_uint(f);
  u += 0x7fffu + ((u >> 16) & 1u);         // round-to-nearest-even
  return (unsigned short)(u >> 16);
}

union Frag {
  v16bf v;
  u32x4 q[2];
  unsigned long long d[4];
  unsigned short u[16];
};

__device__ __forceinline__ v8f wmma_bf16(v16bf a, v16bf b, v8f c) {
  return __builtin_amdgcn_wmma_f32_16x16x32_bf16(false, a, false, b,
                                                 (short)0, c, false, false);
}

// Row reductions across a 16-lane half (C-matrix rows live in lane halves).
__device__ __forceinline__ float rmax16(float v) {
#pragma unroll
  for (int m = 1; m < 16; m <<= 1) v = fmaxf(v, __shfl_xor(v, m, 32));
  return v;
}
__device__ __forceinline__ float rsum16(float v) {
#pragma unroll
  for (int m = 1; m < 16; m <<= 1) v += __shfl_xor(v, m, 32);
  return v;
}

// ---------------------------------------------------------------------------
// Kernel 1: pack weights to bf16 in WMMA-friendly layouts.
//   conv weights (OIHW 128x128x4x4) -> [kchunk(64)][n(128)][kk(32)]
//   Wl3/Wr3 (2048x128) -> same row-major layout, bf16
// ---------------------------------------------------------------------------
__global__ void __launch_bounds__(256) pack_kernel(
    const float* __restrict__ Wl1, const float* __restrict__ Wr1,
    const float* __restrict__ Wl2, const float* __restrict__ Wr2,
    const float* __restrict__ Wl3, const float* __restrict__ Wr3,
    unsigned short* __restrict__ ws) {
  const unsigned idx = blockIdx.x * 256u + threadIdx.x;
  const unsigned g = idx >> 18;
  const unsigned r = idx & 262143u;
  if (g < 4u) {
    const unsigned kc = r >> 12, n = (r >> 5) & 127u, kk = r & 31u;
    const unsigned cin = kc * 2u + (kk >> 4), kp = kk & 15u;
    const float* W = (g == 0u) ? Wl1 : (g == 1u) ? Wr1 : (g == 2u) ? Wl2 : Wr2;
    ws[E_WP + g * 262144u + r] = f2bf(W[(n * 128u + cin) * 16u + kp]);
  } else {
    const float* W = (g == 4u) ? Wl3 : Wr3;
    ws[((g == 4u) ? E_W3L : E_W3R) + r] = f2bf(W[r]);
  }
}

// ---------------------------------------------------------------------------
// Kernel 2: LayerNorm2d over channels, output bf16 NCHW planes.
// One thread owns one (b,h,w) pixel; channel loads coalesce across w.
// ---------------------------------------------------------------------------
__global__ void __launch_bounds__(256) ln_kernel(
    const float* __restrict__ x_l, const float* __restrict__ x_r,
    const float* __restrict__ wl, const float* __restrict__ bl,
    const float* __restrict__ wr, const float* __restrict__ br,
    unsigned short* __restrict__ ws) {
  const int bid = blockIdx.x;
  const int side = bid >> 8;
  const int b = (bid >> 2) & 63, hb = bid & 3;
  const int t = threadIdx.x;
  const int h = hb * 8 + (t >> 5), w = t & 31;
  const float* x = side ? x_r : x_l;
  const float* wgt = side ? wr : wl;
  const float* bp = side ? br : bl;
  unsigned short* o = ws + (side ? E_LNR : E_LNL);
  const size_t base = ((size_t)b << 17) + (size_t)(h * 32 + w);
  float s = 0.f, s2 = 0.f;
  for (int c = 0; c < 128; ++c) {
    const float v = x[base + ((size_t)c << 10)];
    s += v; s2 += v * v;
  }
  const float mu = s * 0.0078125f;
  const float var = s2 * 0.0078125f - mu * mu;
  const float rstd = rsqrtf(var + 1e-6f);
  for (int c = 0; c < 128; ++c) {
    const size_t i = base + ((size_t)c << 10);
    o[i] = f2bf((x[i] - mu) * rstd * wgt[c] + bp[c]);
  }
}

// ---------------------------------------------------------------------------
// Kernel 3: conv4/stride4 as im2col GEMM (M=64 rows per batch, N=128, K=2048).
// g=0: Q_l=ln_l*Wl1+bl1+pos   g=1: Q_r   g=2: V_l=x_l*Wl2+bl2+pos   g=3: V_r
// LDS chunk layout is [m(64)][kk(32)] bf16 (A-fragment order), double
// buffered. Q path (bf16 source): global_load_async_to_lds_b64, one 8B
// kw-run per op, prefetching the next K-chunk under the WMMAs. V path (f32
// source): synchronous load + f32->bf16 convert, same LDS layout.
// Q stored [m][n]; V stored transposed [n][m] for the attention kernel.
// ---------------------------------------------------------------------------
__device__ __forceinline__ void conv_stage(
    int g, int b, int kc, unsigned bufE, int t,
    const unsigned short* __restrict__ srcQ,
    const float* __restrict__ srcV,
    unsigned short* lds) {
  if (g < 2) {
    // Two async 8-byte copies per lane; im2col permutation is in the source
    // address, destination is linear [m][kk].
#pragma unroll
    for (int o = 0; o < 2; ++o) {
      const int e0 = (t * 2 + o) * 4;           // LDS element (4 bf16 run)
      const int m = e0 >> 5, kk0 = e0 & 31;
      const int cl = kk0 >> 4, kh = (kk0 & 15) >> 2;
      const int oh = m >> 3, ow = m & 7;
      const unsigned short* gp = srcQ +
          ((size_t)(b * 128 + kc * 2 + cl) << 10) + (oh * 4 + kh) * 32 + ow * 4;
      const unsigned ldsaddr =
          (unsigned)(unsigned long long)(void*)&lds[bufE + (unsigned)e0];
      asm volatile("global_load_async_to_lds_b64 %0, %1, off"
                   :: "v"(ldsaddr), "v"(gp) : "memory");
    }
  } else {
    // 8 consecutive f32 from one plane row -> two 4-element kw-runs in LDS.
    const int p0 = t * 8;
    const int cl = p0 >> 10, rem = p0 & 1023;
    const int h = rem >> 5, w0 = rem & 31;
    const int oh = h >> 2, kh = h & 3, ow0 = w0 >> 2;
    const float* sp =
        srcV + ((size_t)(b * 128 + kc * 2 + cl) << 10) + h * 32 + w0;
    const f32x4 a0 = *(const f32x4*)sp;
    const f32x4 a1 = *(const f32x4*)(sp + 4);
    const unsigned long long d0 =
        (unsigned long long)f2bf(a0[0]) |
        ((unsigned long long)f2bf(a0[1]) << 16) |
        ((unsigned long long)f2bf(a0[2]) << 32) |
        ((unsigned long long)f2bf(a0[3]) << 48);
    const unsigned long long d1 =
        (unsigned long long)f2bf(a1[0]) |
        ((unsigned long long)f2bf(a1[1]) << 16) |
        ((unsigned long long)f2bf(a1[2]) << 32) |
        ((unsigned long long)f2bf(a1[3]) << 48);
    const int base0 = (oh * 8 + ow0) * 32 + cl * 16 + kh * 4;
    *(unsigned long long*)&lds[bufE + base0] = d0;
    *(unsigned long long*)&lds[bufE + base0 + 32] = d1;   // ow0+1 row
  }
}

__global__ void __launch_bounds__(256) conv_gemm_kernel(
    const float* __restrict__ x_l, const float* __restrict__ x_r,
    const float* __restrict__ bl1, const float* __restrict__ br1,
    const float* __restrict__ bl2, const float* __restrict__ br2,
    const float* __restrict__ pos, unsigned short* __restrict__ ws) {
  __shared__ __align__(16) unsigned short lds[4096];  // 2 x [64][32] bf16
  const int bid = blockIdx.x;
  const int g = bid >> 6;
  const int b = bid & 63;
  const int t = threadIdx.x;
  const int wave = t >> 5, lane = t & 31, ln16 = lane & 15, hi = lane >> 4;

  const unsigned short* wp = ws + E_WP + (unsigned)g * 262144u;
  const float* bias = (g == 0) ? bl1 : (g == 1) ? br1 : (g == 2) ? bl2 : br2;
  const unsigned short* srcQ = ws + ((g == 0) ? E_LNL : E_LNR);
  const float* srcV = (g == 2) ? x_l : x_r;

  const v8f vzero = {};
  v8f acc[4];
#pragma unroll
  for (int i = 0; i < 4; ++i) acc[i] = vzero;

  conv_stage(g, b, 0, 0u, t, srcQ, srcV, lds);  // prologue: chunk 0 -> buf 0

  for (int kc = 0; kc < 64; ++kc) {
    __syncthreads();  // everyone done reading the buffer we are about to fill
    if (kc + 1 < 64)
      conv_stage(g, b, kc + 1, (unsigned)(((kc + 1) & 1) * 2048), t,
                 srcQ, srcV, lds);
    if (g < 2) {
      // 2 async ops per stage: <=2 outstanding means chunk kc has landed.
      if (kc < 63) asm volatile("s_wait_asynccnt 0x2" ::: "memory");
      else         asm volatile("s_wait_asynccnt 0x0" ::: "memory");
    }
    __syncthreads();

    const int bufE = (kc & 1) * 2048;
    // B fragment: 32 contiguous bf16 from packed weights (L2 resident).
    Frag bf;
    {
      const unsigned boff = (unsigned)(kc * 128 + wave * 16 + ln16) * 32u +
                            (unsigned)hi * 8u;
      bf.q[0] = *(const u32x4*)&wp[boff];
      bf.q[1] = *(const u32x4*)&wp[boff + 16];
    }
#pragma unroll
    for (int mt = 0; mt < 4; ++mt) {
      Frag af;  // LDS already in fragment order: two contiguous b128 reads
      const int fo = bufE + (mt * 16 + ln16) * 32 + hi * 8;
      af.q[0] = *(const u32x4*)&lds[fo];
      af.q[1] = *(const u32x4*)&lds[fo + 16];
      acc[mt] = wmma_bf16(af.v, bf.v, acc[mt]);
    }
  }

  const int ncol = wave * 16 + ln16;
  const float bv = bias[ncol];
  unsigned short* outQ = ws + ((g == 0) ? E_QL : E_QR);
  unsigned short* outT = ws + ((g == 2) ? E_VTL : E_VTR);
#pragma unroll
  for (int mt = 0; mt < 4; ++mt) {
#pragma unroll
    for (int r = 0; r < 8; ++r) {
      const int m = mt * 16 + r + hi * 8;
      const unsigned short h = f2bf(acc[mt][r] + bv + pos[ncol * 64 + m]);
      if (g < 2) outQ[(size_t)(b * 64 + m) * 128 + ncol] = h;
      else       outT[(size_t)ncol * 4096 + b * 64 + m] = h;
    }
  }
}

// ---------------------------------------------------------------------------
// Kernel 4: flash cross-attention, both directions in one launch.
// dir0: softmax(Q_l K_r^T * s) V_r -> F_r2l ; dir1: roles swapped -> F_l2r.
// 4 waves x 16 query rows; 32 keys per step; online softmax; P re-laid-out
// through per-wave LDS into an A fragment for the O += P*V WMMA.
// ---------------------------------------------------------------------------
__global__ void __launch_bounds__(128) attn_kernel(unsigned short* __restrict__ ws) {
  __shared__ __align__(16) unsigned short ldsp[2048];  // 4 waves * 16x32 bf16
  const int bid = blockIdx.x;
  const int dir = bid >> 6, qb = bid & 63;
  const int t = threadIdx.x;
  const int wave = t >> 5, lane = t & 31, ln16 = lane & 15, hi = lane >> 4;
  const unsigned short* Q  = ws + ((dir == 0) ? E_QL  : E_QR);
  const unsigned short* K  = ws + ((dir == 0) ? E_QR  : E_QL);
  const unsigned short* Vt = ws + ((dir == 0) ? E_VTR : E_VTL);
  unsigned short* F = ws + ((dir == 0) ? E_FRL : E_FLR);
  const float SCALE = 0.011048543456039806f;  // c^-0.5 * 64^-0.5
  const int qrow = qb * 64 + wave * 16 + ln16;
  const int pbase = wave * 512;

  Frag aq[4];  // cached Q A-fragments (full 128-dim)
#pragma unroll
  for (int kc = 0; kc < 4; ++kc) {
    const size_t base = (size_t)qrow * 128 + kc * 32 + hi * 8;
    aq[kc].q[0] = *(const u32x4*)&Q[base];
    aq[kc].q[1] = *(const u32x4*)&Q[base + 16];
  }

  const v8f vzero = {};
  v8f o[8];
#pragma unroll
  for (int n = 0; n < 8; ++n) o[n] = vzero;
  float mrun[8], lrun[8];
#pragma unroll
  for (int r = 0; r < 8; ++r) { mrun[r] = -3.0e38f; lrun[r] = 0.f; }

  for (int kb = 0; kb < 4096; kb += 32) {
    // S = Q * K^T for 32 keys (two 16-key n-tiles)
    v8f s0 = vzero, s1 = vzero;
#pragma unroll
    for (int kc = 0; kc < 4; ++kc) {
      Frag b0, b1;
      const size_t k0 = (size_t)(kb + ln16) * 128 + kc * 32 + hi * 8;
      b0.q[0] = *(const u32x4*)&K[k0];
      b0.q[1] = *(const u32x4*)&K[k0 + 16];
      const size_t k1 = (size_t)(kb + 16 + ln16) * 128 + kc * 32 + hi * 8;
      b1.q[0] = *(const u32x4*)&K[k1];
      b1.q[1] = *(const u32x4*)&K[k1 + 16];
      s0 = wmma_bf16(aq[kc].v, b0.v, s0);
      s1 = wmma_bf16(aq[kc].v, b1.v, s1);
    }
    // online softmax update; write unnormalized P (bf16) to per-wave LDS
    float alpha[8];
#pragma unroll
    for (int r = 0; r < 8; ++r) {
      const float v0 = s0[r] * SCALE, v1 = s1[r] * SCALE;
      const float tmax = rmax16(fmaxf(v0, v1));
      const float mnew = fmaxf(mrun[r], tmax);
      const float p0 = __expf(v0 - mnew), p1 = __expf(v1 - mnew);
      alpha[r] = __expf(mrun[r] - mnew);
      lrun[r] = lrun[r] * alpha[r] + rsum16(p0 + p1);
      mrun[r] = mnew;
      const int row = r + hi * 8;
      ldsp[pbase + row * 32 + ln16] = f2bf(p0);
      ldsp[pbase + row * 32 + 16 + ln16] = f2bf(p1);
    }
    asm volatile("s_wait_dscnt 0x0" ::: "memory");
#pragma unroll
    for (int n = 0; n < 8; ++n) {
#pragma unroll
      for (int r = 0; r < 8; ++r) o[n][r] *= alpha[r];
    }
    // P as A-fragment (LDS re-layout), V as contiguous B-fragments (Vt)
    Frag pa;
    {
      const int off = pbase + ln16 * 32 + hi * 8;
      pa.q[0] = *(const u32x4*)&ldsp[off];
      pa.q[1] = *(const u32x4*)&ldsp[off + 16];
    }
#pragma unroll
    for (int n = 0; n < 8; ++n) {
      Frag vb;
      const size_t vaddr = (size_t)(n * 16 + ln16) * 4096 + kb + hi * 8;
      vb.q[0] = *(const u32x4*)&Vt[vaddr];
      vb.q[1] = *(const u32x4*)&Vt[vaddr + 16];
      o[n] = wmma_bf16(pa.v, vb.v, o[n]);
    }
  }

#pragma unroll
  for (int r = 0; r < 8; ++r) lrun[r] = 1.0f / lrun[r];
#pragma unroll
  for (int n = 0; n < 8; ++n) {
#pragma unroll
    for (int r = 0; r < 8; ++r) {
      const int row = qb * 64 + wave * 16 + r + hi * 8;
      F[(size_t)row * 128 + n * 16 + ln16] = f2bf(o[n][r] * lrun[r]);
    }
  }
}

// ---------------------------------------------------------------------------
// Kernel 5: F @ W3^T + b3, pixel-shuffle(4), * beta/gamma, + residual x.
// M=64 rows (one batch), N=2048, K=128. 8 waves cycle over the 128 n-tiles.
// ---------------------------------------------------------------------------
__global__ void __launch_bounds__(256) proj_kernel(
    const float* __restrict__ x_l, const float* __restrict__ x_r,
    const float* __restrict__ bl3, const float* __restrict__ br3,
    const float* __restrict__ beta, const float* __restrict__ gamma,
    const unsigned short* __restrict__ ws, float* __restrict__ out) {
  const int bid = blockIdx.x;
  const int side = bid >> 6, b = bid & 63;
  const int t = threadIdx.x;
  const int wave = t >> 5, lane = t & 31, ln16 = lane & 15, hi = lane >> 4;
  const unsigned short* F  = ws + (side ? E_FLR : E_FRL);
  const unsigned short* W3 = ws + (side ? E_W3R : E_W3L);
  const float* bias = side ? br3 : bl3;
  const float* sc   = side ? gamma : beta;
  const float* xin  = side ? x_r : x_l;
  float* op = out + (size_t)side * 8388608u;

  Frag af[4][4];  // F fragments cached for all 64 rows of this batch
#pragma unroll
  for (int mt = 0; mt < 4; ++mt) {
#pragma unroll
    for (int kc = 0; kc < 4; ++kc) {
      const size_t base = (size_t)(b * 64 + mt * 16 + ln16) * 128 + kc * 32 + hi * 8;
      af[mt][kc].q[0] = *(const u32x4*)&F[base];
      af[mt][kc].q[1] = *(const u32x4*)&F[base + 16];
    }
  }

  const v8f vzero = {};
  for (int nt = wave; nt < 128; nt += 8) {
    const int n = nt * 16 + ln16;
    Frag bf[4];
#pragma unroll
    for (int kc = 0; kc < 4; ++kc) {
      const size_t base = (size_t)n * 128 + kc * 32 + hi * 8;
      bf[kc].q[0] = *(const u32x4*)&W3[base];
      bf[kc].q[1] = *(const u32x4*)&W3[base + 16];
    }
    const int c = n >> 4, ij = n & 15, pi = ij >> 2, pj = ij & 3;
    const float bb = bias[n], sv = sc[c];
#pragma unroll
    for (int mt = 0; mt < 4; ++mt) {
      v8f acc = vzero;
#pragma unroll
      for (int kc = 0; kc < 4; ++kc) acc = wmma_bf16(af[mt][kc].v, bf[kc].v, acc);
#pragma unroll
      for (int r = 0; r < 8; ++r) {
        const int m = mt * 16 + r + hi * 8;
        const int oh = m >> 3, ow = m & 7;
        const size_t oaddr =
            ((size_t)(b * 128 + c) * 32 + (size_t)(oh * 4 + pi)) * 32 +
            (size_t)(ow * 4 + pj);
        op[oaddr] = xin[oaddr] + (acc[r] + bb) * sv;
      }
    }
  }
}

// ---------------------------------------------------------------------------
extern "C" void kernel_launch(void* const* d_in, const int* in_sizes, int n_in,
                              void* d_out, int out_size, void* d_ws, size_t ws_size,
                              hipStream_t stream) {
  (void)in_sizes; (void)n_in; (void)out_size; (void)ws_size;
  const float* x_l    = (const float*)d_in[0];
  const float* x_r    = (const float*)d_in[1];
  const float* ln_w_l = (const float*)d_in[2];
  const float* ln_b_l = (const float*)d_in[3];
  const float* ln_w_r = (const float*)d_in[4];
  const float* ln_b_r = (const float*)d_in[5];
  const float* Wl1 = (const float*)d_in[6];
  const float* bl1 = (const float*)d_in[7];
  const float* Wr1 = (const float*)d_in[8];
  const float* br1 = (const float*)d_in[9];
  const float* Wl2 = (const float*)d_in[10];
  const float* bl2 = (const float*)d_in[11];
  const float* Wr2 = (const float*)d_in[12];
  const float* br2 = (const float*)d_in[13];
  const float* pos   = (const float*)d_in[14];
  const float* beta  = (const float*)d_in[15];
  const float* gamma = (const float*)d_in[16];
  const float* Wl3 = (const float*)d_in[17];
  const float* bl3 = (const float*)d_in[18];
  const float* Wr3 = (const float*)d_in[19];
  const float* br3 = (const float*)d_in[20];
  unsigned short* ws = (unsigned short*)d_ws;
  float* out = (float*)d_out;

  pack_kernel<<<6144, 256, 0, stream>>>(Wl1, Wr1, Wl2, Wr2, Wl3, Wr3, ws);
  ln_kernel<<<512, 256, 0, stream>>>(x_l, x_r, ln_w_l, ln_b_l, ln_w_r, ln_b_r, ws);
  conv_gemm_kernel<<<256, 256, 0, stream>>>(x_l, x_r, bl1, br1, bl2, br2, pos, ws);
  attn_kernel<<<128, 128, 0, stream>>>(ws);
  proj_kernel<<<128, 256, 0, stream>>>(x_l, x_r, bl3, br3, beta, gamma, ws, out);
}
